// HPUAttentionImpl_31525059953311
// MI455X (gfx1250) — matrix operand order repaired and checked by reading, not compiled, not run on hardware
//
#include <hip/hip_runtime.h>

// ---------------------------------------------------------------------------
// MI455X (gfx1250) prefill attention: causal + ALiBi + GQA, plus paged-KV
// cache scatter. f16 WMMA (16x16x32) with f32 accumulation; f32 softmax.
// ---------------------------------------------------------------------------

typedef __attribute__((ext_vector_type(16))) _Float16 v16h;
typedef __attribute__((ext_vector_type(8)))  _Float16 v8h;
typedef __attribute__((ext_vector_type(8)))  float    v8f;

#define B_       4
#define S_       1024
#define H_       32
#define HKV_     8
#define D_       128
#define BLK_     128
#define NBLOCKS_ 64
#define SCALE_   0.08838834764831845f   // 1/sqrt(128)
#define LOG2E_   1.4426950408889634f

__device__ __forceinline__ v8f wmma_f32_f16(v16h a, v16h b, v8f c) {
  // D = A(16x32 f16) * B(32x16 f16) + C(16x16 f32)
  return __builtin_amdgcn_wmma_f32_16x16x32_f16(
      /*neg_a=*/false, a, /*neg_b=*/false, b,
      /*c_mod=*/(short)0, c, /*reuse_a=*/false, /*reuse_b=*/false);
}

__global__ __launch_bounds__(256)
void attn_prefill_kernel(const float* __restrict__ q_g,
                         const float* __restrict__ k_g,
                         const float* __restrict__ v_g,
                         const float* __restrict__ slopes_g,
                         float* __restrict__ out_g) {
  constexpr int KPAD = 136;   // halves per kbuf row (128 + 8 pad)
  constexpr int VPAD = 40;    // halves per vt row   (32 + 8 pad)
  constexpr int SROW = 34;    // floats per score row
  constexpr int PROW = 40;    // halves per prob row

  __shared__ _Float16 kbuf[32 * KPAD];          // K tile, [key][dim] f16
  __shared__ _Float16 vtbuf[128 * VPAD];        // V tile transposed, [dim][key] f16
  __shared__ float    sbuf[8][16 * SROW];       // per-wave raw scores
  __shared__ _Float16 pbuf[8][16 * PROW];       // per-wave probabilities (f16)
  __shared__ float    cfac[8][16];              // per-wave row rescale factors

  const int tid    = threadIdx.x;
  const int w      = tid >> 5;
  const int lane   = tid & 31;
  const int col    = lane & 15;
  const bool lohalf = (lane < 16);
  const int hi8    = lohalf ? 0 : 8;

  const int blk  = blockIdx.x;
  const int qblk = blk & 7;              // S/128 = 8 query tiles
  const int h    = (blk >> 3) & 31;      // H = 32
  const int b    = blk >> 8;
  const int hkv  = h >> 2;               // G = 4
  const int q0   = qblk * 128;
  const int qrow0 = q0 + w * 16;         // this wave's 16 query rows
  const float slope = slopes_g[h];

  // ---- Q fragments: A-matrix layout (16x32 f16), 4 K-chunks over D=128 ----
  // Lane half L (0-15): K = 0-15 (V0-7); half H: 16-31.
  v16h qfrag[4];
  {
    const float* qr = q_g + ((size_t)(b * S_ + qrow0 + col) * (H_ * D_) + h * D_);
    const int lo0 = lohalf ? 0 : 16;
#pragma unroll
    for (int kc = 0; kc < 4; ++kc) {
      const int lo = kc * 32 + lo0;
      const int hi = lo + 8;
      v16h a;
#pragma unroll
      for (int i = 0; i < 8; ++i) {
        a[i]     = (_Float16)qr[lo + i];
        a[8 + i] = (_Float16)qr[hi + i];
      }
      qfrag[kc] = a;
    }
  }

  v8f o[8];
#pragma unroll
  for (int n = 0; n < 8; ++n)
#pragma unroll
    for (int r = 0; r < 8; ++r) o[n][r] = 0.0f;

  float m = -1.0e30f, l = 0.0f;

  const int nkt = (q0 >> 5) + 4;         // key tiles of 32 up to causal end
  const int ldflat = tid * 16;           // 4096 f32 per tile / 256 threads
  const int ldrow  = ldflat >> 7;        // key row 0..31
  const int lddim  = ldflat & 127;       // dim 0..112 step 16

  for (int jt = 0; jt < nkt; ++jt) {
    const int kt0 = jt * 32;
    __syncthreads();
    // ---- cooperative stage: K tile (f32->f16) and V tile transposed ----
    {
      const size_t src = (size_t)(b * S_ + kt0 + ldrow) * (HKV_ * D_) + hkv * D_ + lddim;
      const float4* ks = (const float4*)(k_g + src);
      const float4* vs = (const float4*)(v_g + src);
      float4 k0 = ks[0], k1 = ks[1], k2 = ks[2], k3 = ks[3];
      float4 w0 = vs[0], w1 = vs[1], w2 = vs[2], w3 = vs[3];
      v8h hA = { (_Float16)k0.x, (_Float16)k0.y, (_Float16)k0.z, (_Float16)k0.w,
                 (_Float16)k1.x, (_Float16)k1.y, (_Float16)k1.z, (_Float16)k1.w };
      v8h hB = { (_Float16)k2.x, (_Float16)k2.y, (_Float16)k2.z, (_Float16)k2.w,
                 (_Float16)k3.x, (_Float16)k3.y, (_Float16)k3.z, (_Float16)k3.w };
      *(v8h*)&kbuf[ldrow * KPAD + lddim]     = hA;
      *(v8h*)&kbuf[ldrow * KPAD + lddim + 8] = hB;
      float vf[16] = { w0.x, w0.y, w0.z, w0.w, w1.x, w1.y, w1.z, w1.w,
                       w2.x, w2.y, w2.z, w2.w, w3.x, w3.y, w3.z, w3.w };
#pragma unroll
      for (int i = 0; i < 16; ++i)
        vtbuf[(lddim + i) * VPAD + ldrow] = (_Float16)vf[i];
      if (jt + 1 < nkt) {   // lowers to global_prefetch_b8
        __builtin_prefetch(k_g + src + 32 * (HKV_ * D_), 0, 0);
        __builtin_prefetch(v_g + src + 32 * (HKV_ * D_), 0, 0);
      }
    }
    __syncthreads();

    if (kt0 <= qrow0 + 15) {   // skip tiles fully beyond this wave's causal bound
      const int kb = lohalf ? 0 : 16;   // B-layout K offset for this lane half

      // ---- S = Q K^T : two 16-key sub-tiles, 4 K-chunks of 32 dims ----
      v8f acc0, acc1;
#pragma unroll
      for (int r = 0; r < 8; ++r) { acc0[r] = 0.0f; acc1[r] = 0.0f; }
#pragma unroll
      for (int kc = 0; kc < 4; ++kc) {
        const _Float16* p0 = &kbuf[col * KPAD + kc * 32 + kb];
        const _Float16* p1 = &kbuf[(16 + col) * KPAD + kc * 32 + kb];
        v8h b00 = *(const v8h*)p0, b01 = *(const v8h*)(p0 + 8);
        v8h b10 = *(const v8h*)p1, b11 = *(const v8h*)(p1 + 8);
        v16h bf0, bf1;
#pragma unroll
        for (int i = 0; i < 8; ++i) {
          bf0[i] = b00[i]; bf0[8 + i] = b01[i];
          bf1[i] = b10[i]; bf1[8 + i] = b11[i];
        }
        acc0 = wmma_f32_f16(qfrag[kc], bf0, acc0);
        acc1 = wmma_f32_f16(qfrag[kc], bf1, acc1);
      }

      // ---- scale + ALiBi + causal mask; spill scores to per-wave LDS ----
      float* sw = sbuf[w];
#pragma unroll
      for (int r = 0; r < 8; ++r) {
        const int row = r + hi8;
        const int qi  = qrow0 + row;
        const int k0i = kt0 + col;
        const int k1i = k0i + 16;
        const float s0 = (k0i <= qi) ? (acc0[r] * SCALE_ + slope * (float)(k0i - qi)) : -3.0e30f;
        const float s1 = (k1i <= qi) ? (acc1[r] * SCALE_ + slope * (float)(k1i - qi)) : -3.0e30f;
        sw[row * SROW + col]      = s0;
        sw[row * SROW + 16 + col] = s1;
      }
      __builtin_amdgcn_wave_barrier();

      // ---- online softmax: lane owns row=col, one 16-key half ----
      const int half = lane >> 4;
      const float* sr = &sw[col * SROW + half * 16];
      float pm = -3.0e30f;
#pragma unroll
      for (int j = 0; j < 16; ++j) pm = fmaxf(pm, sr[j]);
      pm = fmaxf(pm, __shfl_xor(pm, 16, 32));
      const float mnew = fmaxf(m, pm);
      const float corr = exp2f((m - mnew) * LOG2E_);
      float ps = 0.0f;
      _Float16* pr = &pbuf[w][col * PROW + half * 16];
#pragma unroll
      for (int j = 0; j < 16; ++j) {
        const float p = exp2f((sr[j] - mnew) * LOG2E_);
        ps += p;
        pr[j] = (_Float16)p;
      }
      ps += __shfl_xor(ps, 16, 32);
      l = l * corr + ps;
      m = mnew;
      if (lohalf) cfac[w][col] = corr;
      __builtin_amdgcn_wave_barrier();

      // ---- rescale O; O += P V (8 output 16-dim tiles) ----
      float cf[8];
#pragma unroll
      for (int r = 0; r < 8; ++r) cf[r] = cfac[w][r + hi8];
      v16h pfrag;
      {
        const int lo = lohalf ? 0 : 16;
        const _Float16* pp = &pbuf[w][col * PROW];
#pragma unroll
        for (int i = 0; i < 8; ++i) {
          pfrag[i]     = pp[lo + i];
          pfrag[8 + i] = pp[lo + 8 + i];
        }
      }
#pragma unroll
      for (int n = 0; n < 8; ++n) {
#pragma unroll
        for (int r = 0; r < 8; ++r) o[n][r] *= cf[r];
        const _Float16* vp = &vtbuf[(n * 16 + col) * VPAD + kb];
        v8h bv0 = *(const v8h*)vp, bv1 = *(const v8h*)(vp + 8);
        v16h bf;
#pragma unroll
        for (int i = 0; i < 8; ++i) { bf[i] = bv0[i]; bf[8 + i] = bv1[i]; }
        o[n] = wmma_f32_f16(pfrag, bf, o[n]);
      }
    }
  }

  // ---- normalize by row sum and store ----
  if (lohalf) cfac[w][col] = 1.0f / l;
  __builtin_amdgcn_wave_barrier();
  float inv[8];
#pragma unroll
  for (int r = 0; r < 8; ++r) inv[r] = cfac[w][r + hi8];
#pragma unroll
  for (int n = 0; n < 8; ++n) {
#pragma unroll
    for (int r = 0; r < 8; ++r) {
      const int row = r + hi8;
      out_g[(size_t)(b * S_ + qrow0 + row) * (H_ * D_) + h * D_ + n * 16 + col] =
          o[n][r] * inv[r];
    }
  }
}

// ---------------------------------------------------------------------------
// Paged KV-cache outputs: pass-through copy of input caches, then block
// scatter of the freshly-written K/V (pure bandwidth, ~100 MB total).
// ---------------------------------------------------------------------------
__global__ void cache_copy_kernel(const float4* __restrict__ kin,
                                  const float4* __restrict__ vin,
                                  float4* __restrict__ kout,
                                  float4* __restrict__ vout, int n4) {
  int i = blockIdx.x * blockDim.x + threadIdx.x;
  if (i < n4) { kout[i] = kin[i]; vout[i] = vin[i]; }
}

__global__ void cache_scatter_kernel(const float4* __restrict__ knew,
                                     const float4* __restrict__ vnew,
                                     const int* __restrict__ bidx,
                                     float4* __restrict__ kout,
                                     float4* __restrict__ vout, int total4) {
  constexpr int BLK4 = BLK_ * HKV_ * D_ / 4;   // float4s per cache block
  int i = blockIdx.x * blockDim.x + threadIdx.x;
  if (i >= total4) return;
  int sb  = i / BLK4;
  int off = i - sb * BLK4;
  int db  = bidx[sb];
  kout[(size_t)db * BLK4 + off] = knew[i];
  vout[(size_t)db * BLK4 + off] = vnew[i];
}

// ---------------------------------------------------------------------------
extern "C" void kernel_launch(void* const* d_in, const int* in_sizes, int n_in,
                              void* d_out, int out_size, void* d_ws, size_t ws_size,
                              hipStream_t stream) {
  const float* q    = (const float*)d_in[0];
  const float* k    = (const float*)d_in[1];
  const float* v    = (const float*)d_in[2];
  const float* kc   = (const float*)d_in[3];
  const float* vc   = (const float*)d_in[4];
  const int*   bidx = (const int*)d_in[5];
  const float* sl   = (const float*)d_in[6];
  const int n_used  = in_sizes[5];

  float* out = (float*)d_out;
  float* kco = out + (size_t)B_ * S_ * H_ * D_;
  float* vco = kco + (size_t)NBLOCKS_ * BLK_ * HKV_ * D_;

  const int cache4 = NBLOCKS_ * BLK_ * HKV_ * D_ / 4;
  cache_copy_kernel<<<(cache4 + 255) / 256, 256, 0, stream>>>(
      (const float4*)kc, (const float4*)vc, (float4*)kco, (float4*)vco, cache4);

  const int total4 = n_used * (BLK_ * HKV_ * D_ / 4);
  cache_scatter_kernel<<<(total4 + 255) / 256, 256, 0, stream>>>(
      (const float4*)k, (const float4*)v, bidx, (float4*)kco, (float4*)vco, total4);

  attn_prefill_kernel<<<B_ * H_ * (S_ / 128), 256, 0, stream>>>(q, k, v, sl, out);
}
